// Net_18811956756716
// MI455X (gfx1250) — matrix operand order repaired and checked by reading, not compiled, not run on hardware
//
#include <hip/hip_runtime.h>

// ---------------------------------------------------------------------------
// Types for WMMA bf16: v16bf operands, v8f accumulator (wave32, 16x16x32).
// ---------------------------------------------------------------------------
typedef __attribute__((ext_vector_type(16))) __bf16 v16bf;
typedef __attribute__((ext_vector_type(8)))  float  v8f;

union Frag32B {
    v16bf v;
    uint4 q[2];   // 32 bytes = 16 bf16
};

__device__ __forceinline__ unsigned short f2bf(float f) {
    // round-to-nearest-even f32 -> bf16
    unsigned int u = __float_as_uint(f);
    if ((u & 0x7fffffffu) > 0x7f800000u)          // NaN -> quiet NaN
        return (unsigned short)((u >> 16) | 0x40u);
    return (unsigned short)((u + 0x7fffu + ((u >> 16) & 1u)) >> 16);
}

// ---------------------------------------------------------------------------
// Utility kernels
// ---------------------------------------------------------------------------
__global__ void fill_f32(float* __restrict__ p, float v, long n) {
    long i = (long)blockIdx.x * blockDim.x + threadIdx.x;
    long stride = (long)gridDim.x * blockDim.x;
    for (; i < n; i += stride) p[i] = v;
}

__global__ void deg_scatter(const int* __restrict__ dst, float* __restrict__ deg, int E) {
    int e = blockIdx.x * blockDim.x + threadIdx.x;
    if (e < E) atomicAdd(deg + dst[e], 1.0f);
}

__global__ void rsqrt_inplace(float* __restrict__ d, int n) {
    int i = blockIdx.x * blockDim.x + threadIdx.x;
    if (i < n) d[i] = rsqrtf(fmaxf(d[i], 1.0f));
}

__global__ void cvt_f32_bf16(const float* __restrict__ in, unsigned short* __restrict__ out, long n) {
    long i = (long)blockIdx.x * blockDim.x + threadIdx.x;
    long stride = (long)gridDim.x * blockDim.x;
    for (; i < n; i += stride) out[i] = f2bf(in[i]);
}

// out[i] = bf16(relu(in[i] + bias[i % F])), F must be a power of two (128 here)
__global__ void bias_relu_bf16(const float* __restrict__ in, const float* __restrict__ bias,
                               unsigned short* __restrict__ out, long n, int Fmask) {
    long i = (long)blockIdx.x * blockDim.x + threadIdx.x;
    long stride = (long)gridDim.x * blockDim.x;
    for (; i < n; i += stride) {
        float v = in[i] + bias[(int)(i & Fmask)];
        out[i] = f2bf(fmaxf(v, 0.0f));
    }
}

// ---------------------------------------------------------------------------
// Pack a KxN fp32 weight matrix into v_wmma_f32_16x16x32_bf16 B fragments in this
// code's contiguous k order (matches the A loads): per 32-K-chunk x 16-col tile, lane l<16 holds column
// l with K = kc*32 + 0..15, lane l>=16 holds column l-16 with K = kc*32+16..31.
// Fragments stored contiguously: frag = kc*tilesN + tn, element = lane*16 + i.
// Columns >= N are zero-padded (used for W2: 40 -> 48).
// ---------------------------------------------------------------------------
__global__ void pack_w_bf16(const float* __restrict__ W, unsigned short* __restrict__ Wp,
                            int K, int N, int Npad) {
    int idx = blockIdx.x * blockDim.x + threadIdx.x;
    int total = K * Npad;
    if (idx >= total) return;
    int i    = idx & 15;
    int lane = (idx >> 4) & 31;
    int frag = idx >> 9;                 // / (32*16)
    int tilesN = Npad >> 4;
    int kc = frag / tilesN;
    int tn = frag % tilesN;
    int k  = kc * 32 + ((lane < 16) ? i : (16 + i));
    int n  = tn * 16 + (lane & 15);
    float v = (n < N) ? W[k * N + n] : 0.0f;
    Wp[idx] = f2bf(v);
}

// ---------------------------------------------------------------------------
// Strip GEMM: C[M x (NT*16)] = A[M x K] (bf16 row-major) * Bp (packed frags).
// One wave per 16-row x NT*16-col output strip: the A fragment for each
// 32-wide k-chunk is loaded ONCE and reused across NT WMMAs (distinct
// accumulators -> no WMMA->WMMA RAW hazard, dense issue). W matrix is tiny
// and L0/L2 resident, so B refetch per wave is free; A traffic drops NTx.
// A-fragment layout: lane l<16 = row l, K {0..15};
// lane l>=16 = row l-16, K {16..31}. Two b128 loads per fragment.
// ---------------------------------------------------------------------------
template <int NT>
__global__ __launch_bounds__(128)
void gemm_strip_bf16(const unsigned short* __restrict__ A,
                     const unsigned short* __restrict__ Bp,
                     float* __restrict__ C,
                     int M, int K, int ldc) {
    int wave = blockIdx.x * (blockDim.x >> 5) + (threadIdx.x >> 5);
    int lane = threadIdx.x & 31;
    if (wave >= (M >> 4)) return;                  // wave-uniform exit
    bool lo = lane < 16;

    const unsigned short* Arow = A + (long)(wave * 16 + (lane & 15)) * K;

    const v8f vzero = {0.f, 0.f, 0.f, 0.f, 0.f, 0.f, 0.f, 0.f};
    v8f acc[NT];
    #pragma unroll
    for (int t = 0; t < NT; ++t) acc[t] = vzero;

    int kchunks = K >> 5;
    for (int kc = 0; kc < kchunks; ++kc) {
        Frag32B a;
        int a0 = kc * 32 + (lo ? 0 : 16);
        a.q[0] = *reinterpret_cast<const uint4*>(Arow + a0);
        a.q[1] = *reinterpret_cast<const uint4*>(Arow + a0 + 8);
        #pragma unroll
        for (int t = 0; t < NT; ++t) {
            Frag32B b;
            const unsigned short* bb = Bp + ((long)(kc * NT + t) * 32 + lane) * 16;
            b.q[0] = *reinterpret_cast<const uint4*>(bb);
            b.q[1] = *reinterpret_cast<const uint4*>(bb + 8);
            acc[t] = __builtin_amdgcn_wmma_f32_16x16x32_bf16(
                         /*neg_a=*/false, a.v, /*neg_b=*/false, b.v,
                         /*c_mod=*/(short)0, acc[t],
                         /*reuse_a=*/false, /*reuse_b=*/false);
        }
    }

    // C/D layout: lane l<16 -> N=l, M = base + {0..7}; l>=16 -> N=l-16, M += 8
    int nbase = lane & 15;
    int mbase = wave * 16 + (lo ? 0 : 8);
    #pragma unroll
    for (int t = 0; t < NT; ++t) {
        #pragma unroll
        for (int r = 0; r < 8; ++r)
            C[(long)(mbase + r) * ldc + t * 16 + nbase] = acc[t][r];
    }
}

// ---------------------------------------------------------------------------
// Edge aggregation, F = 128: one wave per edge, each lane a float4 group.
// Edge scalars (src, dst, norm) are wave-uniform -> readfirstlane so the
// compiler can use scalar loads / SALU, keeping VMEM for the 512B gather.
// Items e in [E, E+Nn) are self-loops.
// ---------------------------------------------------------------------------
__global__ void agg128(const float* __restrict__ h, const int* __restrict__ src,
                       const int* __restrict__ dst, const float* __restrict__ dinv,
                       float* __restrict__ out, int E, int Nn) {
    long idx = (long)blockIdx.x * blockDim.x + threadIdx.x;
    long total = (long)(E + Nn) * 32;
    if (idx >= total) return;
    int e = __builtin_amdgcn_readfirstlane((int)(idx >> 5));  // uniform per wave
    int g = (int)(idx & 31);
    int s, d; float nrm;
    if (e < E) { s = src[e]; d = dst[e]; nrm = dinv[s] * dinv[d]; }
    else       { s = d = e - E; float dv = dinv[s]; nrm = dv * dv; }
    const float4 v = *reinterpret_cast<const float4*>(h + (long)s * 128 + g * 4);
    float* o = out + (long)d * 128 + g * 4;
    atomicAdd(o + 0, v.x * nrm);
    atomicAdd(o + 1, v.y * nrm);
    atomicAdd(o + 2, v.z * nrm);
    atomicAdd(o + 3, v.w * nrm);
}

// Edge aggregation, F = 40 (input stride 48 from padded GEMM2, output stride 40).
// 10 threads per edge, each carrying one float4 group (e not wave-uniform here,
// so keep per-lane loads).
__global__ void agg40(const float* __restrict__ h, const int* __restrict__ src,
                      const int* __restrict__ dst, const float* __restrict__ dinv,
                      float* __restrict__ out, int E, int Nn) {
    long idx = (long)blockIdx.x * blockDim.x + threadIdx.x;
    long total = (long)(E + Nn) * 10;          // 10 float4 groups cover 40 floats
    if (idx >= total) return;
    int e = (int)(idx / 10);
    int g = (int)(idx % 10);
    int s, d; float nrm;
    if (e < E) { s = src[e]; d = dst[e]; nrm = dinv[s] * dinv[d]; }
    else       { s = d = e - E; float dv = dinv[s]; nrm = dv * dv; }
    const float4 v = *reinterpret_cast<const float4*>(h + (long)s * 48 + g * 4);
    float* o = out + (long)d * 40 + g * 4;
    atomicAdd(o + 0, v.x * nrm);
    atomicAdd(o + 1, v.y * nrm);
    atomicAdd(o + 2, v.z * nrm);
    atomicAdd(o + 3, v.w * nrm);
}

// In-place: row = log_softmax(row + b2), 40 classes, one thread per node.
__global__ void finalize_logsoftmax(float* __restrict__ out, const float* __restrict__ b2, int Nn) {
    int i = blockIdx.x * blockDim.x + threadIdx.x;
    if (i >= Nn) return;
    float* row = out + (long)i * 40;
    float m = -3.402823466e38f;
    for (int j = 0; j < 40; ++j) m = fmaxf(m, row[j] + b2[j]);
    float s = 0.0f;
    for (int j = 0; j < 40; ++j) s += expf(row[j] + b2[j] - m);
    float ls = logf(s);
    for (int j = 0; j < 40; ++j) row[j] = row[j] + b2[j] - m - ls;
}

// ---------------------------------------------------------------------------
// Launch
// ---------------------------------------------------------------------------
static inline size_t align256(size_t x) { return (x + 255) & ~(size_t)255; }

extern "C" void kernel_launch(void* const* d_in, const int* in_sizes, int n_in,
                              void* d_out, int out_size, void* d_ws, size_t ws_size,
                              hipStream_t stream) {
    const float* x    = (const float*)d_in[0];
    const int*   eidx = (const int*)  d_in[1];
    const float* W1   = (const float*)d_in[2];
    const float* b1   = (const float*)d_in[3];
    const float* W2   = (const float*)d_in[4];
    const float* b2   = (const float*)d_in[5];

    const int Nn = in_sizes[0] / 128;       // 100000 nodes
    const int E  = in_sizes[1] / 2;         // 1600000 edges
    const int*   src = eidx;                // edge_index[0]
    const int*   dst = eidx + E;            // edge_index[1]
    float* outp = (float*)d_out;            // Nn x 40

    // Workspace layout (with reuse across the two layers)
    char* w = (char*)d_ws;
    float*          deg  = (float*)w;          w += align256((size_t)Nn * 4);
    unsigned short* W1p  = (unsigned short*)w; w += align256((size_t)128 * 128 * 2);
    unsigned short* W2p  = (unsigned short*)w; w += align256((size_t)128 * 48 * 2);
    unsigned short* bufA = (unsigned short*)w; w += align256((size_t)Nn * 128 * 2); // x_bf16, then h1_bf16
    float*          bufB = (float*)w;          w += align256((size_t)Nn * 128 * 4); // h (GEMM1), then h2 (GEMM2, ld=48)
    float*          bufC = (float*)w;          /* Nn*128*4 */                        // layer-1 aggregation output

    const int T = 256;
    const long nF1 = (long)Nn * 128;

    // 0) init: deg = 1 (self-loop), zero aggregation buffers
    fill_f32<<<4096, T, 0, stream>>>(deg, 1.0f, Nn);
    fill_f32<<<8192, T, 0, stream>>>(bufC, 0.0f, nF1);
    fill_f32<<<4096, T, 0, stream>>>(outp, 0.0f, (long)Nn * 40);

    // 1) degrees + rsqrt
    deg_scatter<<<(E + T - 1) / T, T, 0, stream>>>(dst, deg, E);
    rsqrt_inplace<<<(Nn + T - 1) / T, T, 0, stream>>>(deg, Nn);

    // 2) operand prep: x -> bf16, pack W1 (128x128), W2 (128x40 -> 128x48)
    cvt_f32_bf16<<<8192, T, 0, stream>>>(x, bufA, nF1);
    pack_w_bf16<<<(128 * 128 + T - 1) / T, T, 0, stream>>>(W1, W1p, 128, 128, 128);
    pack_w_bf16<<<(128 * 48 + T - 1) / T, T, 0, stream>>>(W2, W2p, 128, 40, 48);

    // 3) h = x @ W1  (one wave per 16x128 strip: 6250 waves, 4 waves/block)
    {
        int waves = Nn / 16;
        gemm_strip_bf16<8><<<(waves + 3) / 4, 128, 0, stream>>>(bufA, W1p, bufB, Nn, 128, 128);
    }

    // 4) layer-1 aggregation into bufC
    {
        long items = (long)(E + Nn) * 32;
        agg128<<<(int)((items + T - 1) / T), T, 0, stream>>>(bufB, src, dst, deg, bufC, E, Nn);
    }

    // 5) h1_bf16 = bf16(relu(bufC + b1))   (reuses bufA; x is dead)
    bias_relu_bf16<<<8192, T, 0, stream>>>(bufC, b1, bufA, nF1, 127);

    // 6) h2 = h1 @ W2  (padded N=48 -> NT=3 strip, ldc=48; reuses bufB)
    {
        int waves = Nn / 16;
        gemm_strip_bf16<3><<<(waves + 3) / 4, 128, 0, stream>>>(bufA, W2p, bufB, Nn, 128, 48);
    }

    // 7) layer-2 aggregation directly into d_out (stride 40)
    {
        long items = (long)(E + Nn) * 10;
        agg40<<<(int)((items + T - 1) / T), T, 0, stream>>>(bufB, src, dst, deg, outp, E, Nn);
    }

    // 8) + b2, log_softmax in place
    finalize_logsoftmax<<<(Nn + T - 1) / T, T, 0, stream>>>(outp, b2, Nn);
}
